// Block_64338610094594
// MI455X (gfx1250) — matrix operand order repaired and checked by reading, not compiled, not run on hardware
//
#include <hip/hip_runtime.h>

// ---------------------------------------------------------------------------
// Problem constants (reference: B=4, S=1024, D=2048, H=16, HD=128, FF=5461)
// ---------------------------------------------------------------------------
#define BATCH   4
#define SEQ     1024
#define DMODEL  2048
#define NHEAD   16
#define HDIM    128
#define FF_REAL 5461
#define FF_PAD  5504            // FF padded to a multiple of 64 (and 32 for K)
#define MTOK    (BATCH * SEQ)   // 4096 tokens

typedef __attribute__((ext_vector_type(16))) __bf16 v16bf;
typedef __attribute__((ext_vector_type(8)))  float  v8f;
typedef __attribute__((ext_vector_type(4)))  int    v4i;

struct BPack { uint4 lo, hi; };

static __device__ __forceinline__ v16bf pack16(uint4 lo, uint4 hi) {
  BPack p; p.lo = lo; p.hi = hi;
  return __builtin_bit_cast(v16bf, p);
}
static __device__ __forceinline__ __bf16 f2bf(float f) {
  unsigned u = __builtin_bit_cast(unsigned, f);
  unsigned r = (u + 0x7FFFu + ((u >> 16) & 1u)) >> 16;   // round-to-nearest-even
  return __builtin_bit_cast(__bf16, (unsigned short)r);
}
static __device__ __forceinline__ float bf2f(__bf16 b) {
  unsigned short s = __builtin_bit_cast(unsigned short, b);
  return __builtin_bit_cast(float, (unsigned)s << 16);
}
static __device__ __forceinline__ v8f v8zero() {
  v8f z = {0.f, 0.f, 0.f, 0.f, 0.f, 0.f, 0.f, 0.f};
  return z;
}
static __device__ __forceinline__ v8f wmma_bf16(v16bf a, v16bf b, v8f c) {
  // D = A x B + C, 16x16x32 bf16 -> f32
  return __builtin_amdgcn_wmma_f32_16x16x32_bf16(false, a, false, b, (short)0, c,
                                                 false, false);
}

// ---------------------------------------------------------------------------
// CDNA5 async memory->LDS copy (ASYNCcnt-tracked, no VGPR round trip).
// 16 bytes per lane. Bodies are device-only (AS(3) ptr is 32-bit on gfx1250).
// Builtin wants v4i-typed pointers: global (AS1) src, LDS (AS3) dst.
// ---------------------------------------------------------------------------
static __device__ __forceinline__ void async_cp16(const __bf16* g, __bf16* l) {
#if defined(__HIP_DEVICE_COMPILE__)
#if __has_builtin(__builtin_amdgcn_global_load_async_to_lds_b128)
  typedef __attribute__((address_space(1))) v4i* gv4i;
  typedef __attribute__((address_space(3))) v4i* lv4i;
  gv4i gp = (gv4i)(unsigned long long)(uintptr_t)g;
  lv4i lp = (lv4i)(unsigned)(uintptr_t)l;   // low 32 bits = LDS offset
  __builtin_amdgcn_global_load_async_to_lds_b128(gp, lp, 0, 0);
#else
  asm volatile("global_load_async_to_lds_b128 %0, %1, off"
               :: "v"((unsigned)(uintptr_t)l), "v"(g)
               : "memory");
#endif
#endif
}
static __device__ __forceinline__ void wait_async0() {
#if defined(__HIP_DEVICE_COMPILE__)
#if __has_builtin(__builtin_amdgcn_s_wait_asynccnt)
  __builtin_amdgcn_s_wait_asynccnt(0);
#else
  asm volatile("s_wait_asynccnt 0x0" ::: "memory");
#endif
#endif
}

// ---------------------------------------------------------------------------
// Weight transpose + f32 -> bf16 convert (Wt[n][k] = W[k][n], zero padded)
// ---------------------------------------------------------------------------
__global__ void transpose_cvt_kernel(const float* __restrict__ W,
                                     __bf16* __restrict__ Wt,
                                     int K, int N, int Kp, int Np) {
  size_t i = (size_t)blockIdx.x * blockDim.x + threadIdx.x;
  size_t total = (size_t)Np * Kp;
  if (i >= total) return;
  int kk = (int)(i % Kp);
  int nn = (int)(i / Kp);
  float v = (kk < K && nn < N) ? W[(size_t)kk * N + nn] : 0.f;
  Wt[i] = f2bf(v);
}

// ---------------------------------------------------------------------------
// RMSNorm over Dn, output bf16 (used for ln1 and ln2)
// ---------------------------------------------------------------------------
__global__ __launch_bounds__(256) void rmsnorm_kernel(const float* __restrict__ x,
                                                      const float* __restrict__ w,
                                                      __bf16* __restrict__ out,
                                                      int Dn) {
  __shared__ float red[256];
  size_t tok = blockIdx.x;
  int t = threadIdx.x;
  const float* xp = x + tok * Dn;
  float ss = 0.f;
  for (int i = t; i < Dn; i += 256) { float v = xp[i]; ss += v * v; }
  red[t] = ss; __syncthreads();
  for (int o = 128; o > 0; o >>= 1) { if (t < o) red[t] += red[t + o]; __syncthreads(); }
  float sc = rsqrtf(red[0] / (float)Dn + 1e-5f);
  __bf16* op = out + tok * Dn;
  for (int i = t; i < Dn; i += 256) op[i] = f2bf(xp[i] * sc * w[i]);
}

// ---------------------------------------------------------------------------
// Per-head qk-RMSNorm over HD=128, f32 in -> bf16 out. blockIdx = token*H + h
// ---------------------------------------------------------------------------
__global__ __launch_bounds__(128) void qknorm_kernel(const float* __restrict__ q,
                                                     const float* __restrict__ w,
                                                     __bf16* __restrict__ out) {
  __shared__ float red[128];
  size_t idx = blockIdx.x;
  int d = threadIdx.x;
  float v = q[idx * HDIM + d];
  red[d] = v * v; __syncthreads();
  for (int o = 64; o > 0; o >>= 1) { if (d < o) red[d] += red[d + o]; __syncthreads(); }
  float sc = rsqrtf(red[0] / (float)HDIM + 1e-5f);
  out[idx * HDIM + d] = f2bf(v * sc * w[d]);
}

// ---------------------------------------------------------------------------
// Forget gate: logf[tok][h] = logsigmoid(h_tok . wf[:,h] + bf[h]).
// One wave per head (16 waves / block), one block per token.
// ---------------------------------------------------------------------------
__global__ __launch_bounds__(512) void fgate_kernel(const __bf16* __restrict__ hb,
                                                    const float* __restrict__ wf,
                                                    const float* __restrict__ bfv,
                                                    float* __restrict__ lf) {
  size_t tok = blockIdx.x;
  int wave = threadIdx.x >> 5, lane = threadIdx.x & 31;
  const __bf16* hp = hb + tok * DMODEL;
  float s = 0.f;
  for (int d = lane; d < DMODEL; d += 32) s += bf2f(hp[d]) * wf[(size_t)d * NHEAD + wave];
  for (int o = 16; o > 0; o >>= 1) s += __shfl_xor(s, o, 32);
  if (lane == 0) {
    float z = s + bfv[wave];
    lf[tok * NHEAD + wave] = fminf(z, 0.f) - log1pf(__expf(-fabsf(z)));
  }
}

// ---------------------------------------------------------------------------
// Inclusive scan over the sequence per (b,h): c[b][h][s] = sum_{t<=s} logf.
// ---------------------------------------------------------------------------
__global__ __launch_bounds__(1024) void cumsum_kernel(const float* __restrict__ lf,
                                                      float* __restrict__ c) {
  __shared__ float buf[SEQ];
  int bh = blockIdx.x;
  int b = bh >> 4, h = bh & 15;
  int s = threadIdx.x;
  buf[s] = lf[((size_t)b * SEQ + s) * NHEAD + h];
  __syncthreads();
  for (int o = 1; o < SEQ; o <<= 1) {
    float v = buf[s];
    float a = (s >= o) ? buf[s - o] : 0.f;
    __syncthreads();
    buf[s] = v + a;
    __syncthreads();
  }
  c[(size_t)bh * SEQ + s] = buf[s];
}

// ---------------------------------------------------------------------------
// V transpose for attention: vt[b][h][d][s] = v[b][s][h][d]
// ---------------------------------------------------------------------------
__global__ void vtrans_kernel(const __bf16* __restrict__ v, __bf16* __restrict__ vt) {
  size_t i = (size_t)blockIdx.x * blockDim.x + threadIdx.x;   // < MTOK*DMODEL
  int d = (int)(i % HDIM);
  int h = (int)((i / HDIM) % NHEAD);
  size_t tok = i / ((size_t)HDIM * NHEAD);
  int s = (int)(tok % SEQ);
  int b = (int)(tok / SEQ);
  vt[(((size_t)b * NHEAD + h) * HDIM + d) * SEQ + s] = v[i];
}

// ---------------------------------------------------------------------------
// WMMA bf16 GEMM: C[M,N] = A[M,K] * B[K,N], B given transposed (Bt[n][k]).
// 128x64 tile / 256-thread block, BK = 32, 8 waves each owning 32x32.
// Double-buffered LDS, staged via async global->LDS (ASYNCcnt), one barrier
// per k-step; next tile streams while current tile's WMMAs execute.
// ---------------------------------------------------------------------------
__global__ __launch_bounds__(256) void gemm_bf16_kernel(
    const __bf16* __restrict__ A, const __bf16* __restrict__ Bt,
    const float* __restrict__ resid, float* __restrict__ outF,
    __bf16* __restrict__ outB, int M, int N, int K) {
  __shared__ __bf16 As[2][128 * 40];   // +8 pad keeps 16B align, spreads banks
  __shared__ __bf16 Bs[2][64 * 40];
  const int n0 = blockIdx.x * 64, m0 = blockIdx.y * 128;
  const int tid = threadIdx.x;
  const int lane = tid & 31, wave = tid >> 5;
  const int wm = wave >> 1, wn = wave & 1;
  const int col = lane & 15, half = lane >> 4;
  const int hi8 = half * 8, kb16 = half * 16;

  // fixed per-thread staging assignment (2 A chunks + 1 B chunk, 16B each)
  const int ar0 = tid >> 2, ac = (tid & 3) << 3;       // A rows 0..63
  const int ar1 = ar0 + 64;                            // A rows 64..127
  const int br  = tid >> 2, bc = ac;                   // B rows 0..63

  v8f acc[2][2];
  for (int i = 0; i < 2; ++i)
    for (int j = 0; j < 2; ++j) acc[i][j] = v8zero();

  const int nk = K >> 5;
  // prologue: stage tile 0
  async_cp16(A + (size_t)(m0 + ar0) * K + ac, &As[0][ar0 * 40 + ac]);
  async_cp16(A + (size_t)(m0 + ar1) * K + ac, &As[0][ar1 * 40 + ac]);
  async_cp16(Bt + (size_t)(n0 + br) * K + bc, &Bs[0][br * 40 + bc]);

  for (int ki = 0; ki < nk; ++ki) {
    wait_async0();        // my tile-ki writes have landed in LDS
    __syncthreads();      // everyone's have; previous buffer fully consumed
    if (ki + 1 < nk) {    // stream tile ki+1 while computing tile ki
      int k0 = (ki + 1) << 5;
      __bf16* Anext = As[(ki + 1) & 1];
      __bf16* Bnext = Bs[(ki + 1) & 1];
      async_cp16(A + (size_t)(m0 + ar0) * K + k0 + ac, &Anext[ar0 * 40 + ac]);
      async_cp16(A + (size_t)(m0 + ar1) * K + k0 + ac, &Anext[ar1 * 40 + ac]);
      async_cp16(Bt + (size_t)(n0 + br) * K + k0 + bc, &Bnext[br * 40 + bc]);
    }
    const __bf16* Ab = As[ki & 1];
    const __bf16* Bb = Bs[ki & 1];
    v16bf af[2], bfr[2];
    for (int ti = 0; ti < 2; ++ti) {
      const __bf16* p = &Ab[(wm * 32 + ti * 16 + col) * 40];
      af[ti] = pack16(*(const uint4*)(p + hi8), *(const uint4*)(p + 16 + hi8));
    }
    for (int tj = 0; tj < 2; ++tj) {
      const __bf16* p = &Bb[(wn * 32 + tj * 16 + col) * 40 + hi8];
      bfr[tj] = pack16(*(const uint4*)p, *(const uint4*)(p + 16));
    }
    for (int ti = 0; ti < 2; ++ti)
      for (int tj = 0; tj < 2; ++tj)
        acc[ti][tj] = wmma_bf16(af[ti], bfr[tj], acc[ti][tj]);
  }

  for (int ti = 0; ti < 2; ++ti)
    for (int tj = 0; tj < 2; ++tj)
      for (int r = 0; r < 8; ++r) {
        int gm = m0 + wm * 32 + ti * 16 + r + hi8;
        int gn = n0 + wn * 32 + tj * 16 + col;
        size_t idx = (size_t)gm * N + gn;
        float v = acc[ti][tj][r];
        if (resid) v += resid[idx];
        if (outF) outF[idx] = v;
        else      outB[idx] = f2bf(v);
      }
}

// ---------------------------------------------------------------------------
// Fused SwiGLU gate GEMM: act = silu(A*W1) .* (A*W2), both weights transposed.
// Same async double-buffered pipeline as gemm_bf16_kernel.
// ---------------------------------------------------------------------------
__global__ __launch_bounds__(256) void mlp_gate_kernel(
    const __bf16* __restrict__ A, const __bf16* __restrict__ B1t,
    const __bf16* __restrict__ B2t, __bf16* __restrict__ outAct,
    int M, int N, int K) {
  __shared__ __bf16 As[2][128 * 40];
  __shared__ __bf16 B1s[2][64 * 40];
  __shared__ __bf16 B2s[2][64 * 40];
  const int n0 = blockIdx.x * 64, m0 = blockIdx.y * 128;
  const int tid = threadIdx.x;
  const int lane = tid & 31, wave = tid >> 5;
  const int wm = wave >> 1, wn = wave & 1;
  const int col = lane & 15, half = lane >> 4;
  const int hi8 = half * 8, kb16 = half * 16;

  const int ar0 = tid >> 2, ac = (tid & 3) << 3;
  const int ar1 = ar0 + 64;
  const int br  = tid >> 2, bc = ac;

  v8f acc1[2][2], acc2[2][2];
  for (int i = 0; i < 2; ++i)
    for (int j = 0; j < 2; ++j) { acc1[i][j] = v8zero(); acc2[i][j] = v8zero(); }

  const int nk = K >> 5;
  async_cp16(A + (size_t)(m0 + ar0) * K + ac, &As[0][ar0 * 40 + ac]);
  async_cp16(A + (size_t)(m0 + ar1) * K + ac, &As[0][ar1 * 40 + ac]);
  async_cp16(B1t + (size_t)(n0 + br) * K + bc, &B1s[0][br * 40 + bc]);
  async_cp16(B2t + (size_t)(n0 + br) * K + bc, &B2s[0][br * 40 + bc]);

  for (int ki = 0; ki < nk; ++ki) {
    wait_async0();
    __syncthreads();
    if (ki + 1 < nk) {
      int k0 = (ki + 1) << 5;
      __bf16* Anext  = As[(ki + 1) & 1];
      __bf16* B1next = B1s[(ki + 1) & 1];
      __bf16* B2next = B2s[(ki + 1) & 1];
      async_cp16(A + (size_t)(m0 + ar0) * K + k0 + ac, &Anext[ar0 * 40 + ac]);
      async_cp16(A + (size_t)(m0 + ar1) * K + k0 + ac, &Anext[ar1 * 40 + ac]);
      async_cp16(B1t + (size_t)(n0 + br) * K + k0 + bc, &B1next[br * 40 + bc]);
      async_cp16(B2t + (size_t)(n0 + br) * K + k0 + bc, &B2next[br * 40 + bc]);
    }
    const __bf16* Ab  = As[ki & 1];
    const __bf16* B1b = B1s[ki & 1];
    const __bf16* B2b = B2s[ki & 1];
    v16bf af[2], b1f[2], b2f[2];
    for (int ti = 0; ti < 2; ++ti) {
      const __bf16* p = &Ab[(wm * 32 + ti * 16 + col) * 40];
      af[ti] = pack16(*(const uint4*)(p + hi8), *(const uint4*)(p + 16 + hi8));
    }
    for (int tj = 0; tj < 2; ++tj) {
      const __bf16* p1 = &B1b[(wn * 32 + tj * 16 + col) * 40 + hi8];
      const __bf16* p2 = &B2b[(wn * 32 + tj * 16 + col) * 40 + hi8];
      b1f[tj] = pack16(*(const uint4*)p1, *(const uint4*)(p1 + 16));
      b2f[tj] = pack16(*(const uint4*)p2, *(const uint4*)(p2 + 16));
    }
    for (int ti = 0; ti < 2; ++ti)
      for (int tj = 0; tj < 2; ++tj) {
        acc1[ti][tj] = wmma_bf16(af[ti], b1f[tj], acc1[ti][tj]);
        acc2[ti][tj] = wmma_bf16(af[ti], b2f[tj], acc2[ti][tj]);
      }
  }

  for (int ti = 0; ti < 2; ++ti)
    for (int tj = 0; tj < 2; ++tj)
      for (int r = 0; r < 8; ++r) {
        int gm = m0 + wm * 32 + ti * 16 + r + hi8;
        int gn = n0 + wn * 32 + tj * 16 + col;
        float g1 = acc1[ti][tj][r];
        float g2 = acc2[ti][tj][r];
        float u = (g1 / (1.f + __expf(-g1))) * g2;   // silu(g1)*g2
        outAct[(size_t)gm * N + gn] = f2bf(u);
      }
}

// ---------------------------------------------------------------------------
// Forgetting flash attention. One wave per (b, h, 16-query tile).
// S^T = K * Q^T so softmaxed P lands directly in WMMA A-fragment layout.
// Decay: softmax_j(qk/sqrt(d) + c_i - c_j) == softmax_j(qk/sqrt(d) - c_j).
// ---------------------------------------------------------------------------
__global__ __launch_bounds__(256) void attn_kernel(
    const __bf16* __restrict__ q, const __bf16* __restrict__ k,
    const __bf16* __restrict__ vt, const float* __restrict__ cdec,
    __bf16* __restrict__ o) {
  const int wid  = blockIdx.x * (blockDim.x >> 5) + (threadIdx.x >> 5);
  const int lane = threadIdx.x & 31;
  const int col  = lane & 15;
  const int half = lane >> 4;
  const int hi8  = half * 8;
  const int kb16 = half * 16;

  const int qt = wid & 63;           // S/16 tiles
  const int h  = (wid >> 6) & 15;
  const int b  = wid >> 10;
  const int qbase = qt * 16;

  const float scale = 0.0883883476483184f;   // 1/sqrt(128)
  const float* cb = cdec + ((size_t)b * NHEAD + h) * SEQ;

  // Q fragments (B operand), resident for the whole wave.
  v16bf qf[4];
  {
    const __bf16* qp = q + (((size_t)(b * SEQ + qbase + col)) * NHEAD + h) * HDIM;
    for (int dc = 0; dc < 4; ++dc) {
      const uint4* p = (const uint4*)(qp + dc * 32 + hi8);
      qf[dc] = pack16(p[0], p[2]);
    }
  }

  v8f oacc[8];
  for (int i = 0; i < 8; ++i) oacc[i] = v8zero();
  float m_run = -1e30f, l_run = 0.f;

  const int nkb = (qbase + 16 + 31) >> 5;    // 32-key blocks (causal)
  for (int kbi = 0; kbi < nkb; ++kbi) {
    const int kb0 = kbi << 5;
    float sc16[16];
    for (int kt = 0; kt < 2; ++kt) {
      v8f st = v8zero();
      int keyrow = kb0 + kt * 16 + col;
      int keyc = keyrow < SEQ ? keyrow : SEQ - 1;
      const __bf16* kp = k + (((size_t)(b * SEQ + keyc)) * NHEAD + h) * HDIM;
      for (int dc = 0; dc < 4; ++dc) {
        const __bf16* ka = kp + dc * 32;
        uint4 a0 = *(const uint4*)(ka + hi8);
        uint4 a1 = *(const uint4*)(ka + 16 + hi8);
        st = wmma_bf16(pack16(a0, a1), qf[dc], st);   // S^T tile
      }
      for (int r = 0; r < 8; ++r) {
        int key = kb0 + kt * 16 + r + hi8;
        int keyi = key < SEQ ? key : SEQ - 1;
        float sv = st[r] * scale - cb[keyi];
        sc16[kt * 8 + r] = (key <= qbase + col) ? sv : -1e30f;
      }
    }
    // online softmax per query (query = lane&15; partner half via shfl_xor 16)
    float tmax = -1e30f;
    for (int i = 0; i < 16; ++i) tmax = fmaxf(tmax, sc16[i]);
    tmax = fmaxf(tmax, __shfl_xor(tmax, 16, 32));
    float mnew = fmaxf(m_run, tmax);
    float alpha = __expf(m_run - mnew);
    float rsum = 0.f;
    v16bf pf;
    for (int i = 0; i < 16; ++i) {
      float p = __expf(sc16[i] - mnew);
      rsum += p;
      pf[i] = f2bf(p);           // exactly the A-fragment element order
    }
    rsum += __shfl_xor(rsum, 16, 32);
    l_run = l_run * alpha + rsum;
    m_run = mnew;

    float fr[8];
    for (int r = 0; r < 8; ++r) fr[r] = __shfl(alpha, r + hi8, 32);
    int vs = kb0 + hi8; if (vs > SEQ - 24) vs = SEQ - 24;   // safe (P==0 there)
    for (int nt = 0; nt < 8; ++nt) {
      for (int r = 0; r < 8; ++r) oacc[nt][r] *= fr[r];
      const __bf16* vp =
          vt + (((size_t)(b * NHEAD + h)) * HDIM + nt * 16 + col) * SEQ + vs;
      const uint4* pv = (const uint4*)vp;
      oacc[nt] = wmma_bf16(pf, pack16(pv[0], pv[2]), oacc[nt]);
    }
  }

  float lr[8];
  for (int r = 0; r < 8; ++r) lr[r] = __shfl(l_run, r + hi8, 32);
  for (int nt = 0; nt < 8; ++nt)
    for (int r = 0; r < 8; ++r) {
      int row = qbase + r + hi8;
      o[(((size_t)(b * SEQ + row)) * NHEAD + h) * HDIM + nt * 16 + col] =
          f2bf(oacc[nt][r] / lr[r]);
    }
}

// ---------------------------------------------------------------------------
// Launcher
// ---------------------------------------------------------------------------
extern "C" void kernel_launch(void* const* d_in, const int* in_sizes, int n_in,
                              void* d_out, int out_size, void* d_ws, size_t ws_size,
                              hipStream_t stream) {
  (void)in_sizes; (void)n_in; (void)out_size; (void)ws_size;
  const float* x    = (const float*)d_in[0];
  const float* ln1w = (const float*)d_in[1];
  const float* ln2w = (const float*)d_in[2];
  const float* wq   = (const float*)d_in[3];
  const float* wk   = (const float*)d_in[4];
  const float* wv   = (const float*)d_in[5];
  const float* wfw  = (const float*)d_in[6];
  const float* bfv  = (const float*)d_in[7];
  const float* qnw  = (const float*)d_in[8];
  const float* knw  = (const float*)d_in[9];
  const float* wo   = (const float*)d_in[10];
  const float* w1   = (const float*)d_in[11];
  const float* w2   = (const float*)d_in[12];
  const float* w3   = (const float*)d_in[13];
  float* out = (float*)d_out;

  char* ws = (char*)d_ws;
  size_t off = 0;
  auto take = [&](size_t bytes) -> void* {
    void* p = ws + off;
    off += (bytes + 255) & ~(size_t)255;
    return p;
  };
  __bf16* h_bf = (__bf16*)take((size_t)MTOK * DMODEL * 2);
  __bf16* wqT  = (__bf16*)take((size_t)DMODEL * DMODEL * 2);
  __bf16* wkT  = (__bf16*)take((size_t)DMODEL * DMODEL * 2);
  __bf16* wvT  = (__bf16*)take((size_t)DMODEL * DMODEL * 2);
  __bf16* woT  = (__bf16*)take((size_t)DMODEL * DMODEL * 2);
  __bf16* w1T  = (__bf16*)take((size_t)FF_PAD * DMODEL * 2);
  __bf16* w2T  = (__bf16*)take((size_t)FF_PAD * DMODEL * 2);
  __bf16* w3T  = (__bf16*)take((size_t)DMODEL * FF_PAD * 2);
  float*  qf   = (float*) take((size_t)MTOK * DMODEL * 4);
  float*  kf   = (float*) take((size_t)MTOK * DMODEL * 4);
  __bf16* vb   = (__bf16*)take((size_t)MTOK * DMODEL * 2);
  __bf16* qb   = (__bf16*)take((size_t)MTOK * DMODEL * 2);
  __bf16* kbuf = (__bf16*)take((size_t)MTOK * DMODEL * 2);
  __bf16* vt   = (__bf16*)take((size_t)MTOK * DMODEL * 2);
  float*  lf   = (float*) take((size_t)MTOK * NHEAD * 4);
  float*  cdec = (float*) take((size_t)MTOK * NHEAD * 4);
  float*  x1   = (float*) take((size_t)MTOK * DMODEL * 4);
  __bf16* h2b  = (__bf16*)take((size_t)MTOK * DMODEL * 2);
  // lifetime-safe aliases: h is dead after fgate/QKV; q/k f32 dead after qknorm
  __bf16* ob  = h_bf;                 // attention output
  __bf16* act = (__bf16*)qf;          // [MTOK, FF_PAD] bf16 fits in qf+kf region

  auto launch_tc = [&](const float* W, __bf16* Wt, int K, int N, int Kp, int Np) {
    size_t total = (size_t)Np * Kp;
    transpose_cvt_kernel<<<(unsigned)((total + 255) / 256), 256, 0, stream>>>(
        W, Wt, K, N, Kp, Np);
  };
  launch_tc(wq, wqT, DMODEL, DMODEL, DMODEL, DMODEL);
  launch_tc(wk, wkT, DMODEL, DMODEL, DMODEL, DMODEL);
  launch_tc(wv, wvT, DMODEL, DMODEL, DMODEL, DMODEL);
  launch_tc(wo, woT, DMODEL, DMODEL, DMODEL, DMODEL);
  launch_tc(w1, w1T, DMODEL, FF_REAL, DMODEL, FF_PAD);
  launch_tc(w2, w2T, DMODEL, FF_REAL, DMODEL, FF_PAD);
  launch_tc(w3, w3T, FF_REAL, DMODEL, FF_PAD, DMODEL);

  rmsnorm_kernel<<<MTOK, 256, 0, stream>>>(x, ln1w, h_bf, DMODEL);

  dim3 gq(DMODEL / 64, MTOK / 128);
  gemm_bf16_kernel<<<gq, 256, 0, stream>>>(h_bf, wqT, nullptr, qf, nullptr,
                                           MTOK, DMODEL, DMODEL);
  gemm_bf16_kernel<<<gq, 256, 0, stream>>>(h_bf, wkT, nullptr, kf, nullptr,
                                           MTOK, DMODEL, DMODEL);
  gemm_bf16_kernel<<<gq, 256, 0, stream>>>(h_bf, wvT, nullptr, nullptr, vb,
                                           MTOK, DMODEL, DMODEL);

  fgate_kernel<<<MTOK, 512, 0, stream>>>(h_bf, wfw, bfv, lf);
  cumsum_kernel<<<BATCH * NHEAD, SEQ, 0, stream>>>(lf, cdec);

  qknorm_kernel<<<MTOK * NHEAD, 128, 0, stream>>>(qf, qnw, qb);
  qknorm_kernel<<<MTOK * NHEAD, 128, 0, stream>>>(kf, knw, kbuf);
  vtrans_kernel<<<(MTOK * DMODEL) / 256, 256, 0, stream>>>(vb, vt);

  attn_kernel<<<(BATCH * NHEAD * (SEQ / 16)) / 8, 256, 0, stream>>>(qb, kbuf, vt,
                                                                    cdec, ob);

  gemm_bf16_kernel<<<gq, 256, 0, stream>>>(ob, woT, x, x1, nullptr,
                                           MTOK, DMODEL, DMODEL);

  rmsnorm_kernel<<<MTOK, 256, 0, stream>>>(x1, ln2w, h2b, DMODEL);

  dim3 gmlp(FF_PAD / 64, MTOK / 128);
  mlp_gate_kernel<<<gmlp, 256, 0, stream>>>(h2b, w1T, w2T, act, MTOK, FF_PAD, DMODEL);

  gemm_bf16_kernel<<<gq, 256, 0, stream>>>(act, w3T, x1, out, nullptr,
                                           MTOK, DMODEL, FF_PAD);
}
